// SerriformBlock_88716844466865
// MI455X (gfx1250) — compile-verified, matrix operand order checked by reading, not run on hardware
//
#include <hip/hip_runtime.h>

// ---------------------------------------------------------------------------
// SerriformBlock for MI455X (gfx1250, wave32, WMMA).
// GEMMs (120 GFLOP) run as bf16 v_wmma_f32_16x16x32_bf16 with LDS double
// buffering; tile staging uses GLOBAL_LOAD_ASYNC_TO_LDS_B128 (ASYNCcnt) when
// the toolchain exposes it; everything elementwise is fused into GEMM
// epilogues; the EMA scan is a 3-pass chunked parallel prefix.
// ---------------------------------------------------------------------------

typedef __attribute__((ext_vector_type(16))) __bf16 v16bf;
typedef __attribute__((ext_vector_type(8)))  __bf16 v8bf;
typedef __attribute__((ext_vector_type(8)))  float  v8f;
typedef __attribute__((ext_vector_type(4)))  int    v4i;

static constexpr int MROWS = 8 * 2048;   // B*L = 16384
static constexpr int DIMC  = 1024;
static constexpr int ACTC  = 256;
static constexpr int LSEQ  = 2048;
static constexpr int NCHUNK = 16;        // scan chunks per sequence
static constexpr int TCHUNK = LSEQ / NCHUNK; // 128

#if defined(__has_builtin)
#if __has_builtin(__builtin_amdgcn_global_load_async_to_lds_b128) && \
    __has_builtin(__builtin_amdgcn_s_wait_asynccnt)
#define USE_ASYNC_LDS 1
#endif
#endif
#ifndef USE_ASYNC_LDS
#define USE_ASYNC_LDS 0
#endif

#define AS1 __attribute__((address_space(1)))
#define AS3 __attribute__((address_space(3)))

// ---------------------------- small kernels --------------------------------

__global__ __launch_bounds__(256) void cvt_bf16_kernel(
    const float* __restrict__ in, __bf16* __restrict__ out, int n) {
  int i = blockIdx.x * 256 + threadIdx.x;
  if (i < n) out[i] = (__bf16)in[i];
}

__global__ __launch_bounds__(256) void rmsnorm_kernel(
    const float* __restrict__ x, const float* __restrict__ nw,
    float* __restrict__ xn) {
  const int row = blockIdx.x;
  const size_t base = (size_t)row * DIMC;
  const int tid = threadIdx.x;
  float v[4];
  float ss = 0.f;
#pragma unroll
  for (int i = 0; i < 4; ++i) {
    v[i] = x[base + tid + i * 256];
    ss += v[i] * v[i];
  }
#pragma unroll
  for (int off = 16; off > 0; off >>= 1) ss += __shfl_xor(ss, off, 32);
  __shared__ float red[8];
  if ((tid & 31) == 0) red[tid >> 5] = ss;
  __syncthreads();
  if (tid == 0) {
    float t = 0.f;
#pragma unroll
    for (int i = 0; i < 8; ++i) t += red[i];
    red[0] = t;
  }
  __syncthreads();
  const float inv = rsqrtf(red[0] * (1.0f / 1024.0f) + 1e-6f);
#pragma unroll
  for (int i = 0; i < 4; ++i) {
    const int c = tid + i * 256;
    xn[base + c] = nw[c] * v[i] * inv;
  }
}

__global__ __launch_bounds__(256) void dwconv_kernel(
    const float* __restrict__ xn, const float* __restrict__ w,
    const float* __restrict__ b, __bf16* __restrict__ xc) {
  const int m = blockIdx.x;          // token row 0..16383
  const int l = m & (LSEQ - 1);
  const int tid = threadIdx.x;
#pragma unroll
  for (int i = 0; i < 4; ++i) {
    const int d = tid + i * 256;
    float acc = b[d];
#pragma unroll
    for (int k = 0; k < 5; ++k) {
      const int li = l + k - 2;
      if (li >= 0 && li < LSEQ)
        acc += xn[(size_t)(m + k - 2) * DIMC + d] * w[d * 5 + k];
    }
    xc[(size_t)m * DIMC + d] = (__bf16)acc;
  }
}

// --- 3-pass chunked scan: h_l = a*h_{l-1} + be*x_l over L per (b, channel) ---

__global__ __launch_bounds__(256) void scan_chunk_kernel(
    const float* __restrict__ xn, const float* __restrict__ alpha,
    const float* __restrict__ beta, float* __restrict__ hs,
    float* __restrict__ carryB) {
  const int b = blockIdx.x >> 4;
  const int ch = blockIdx.x & (NCHUNK - 1);
  const int c = threadIdx.x;
  const float a = alpha[c], be = beta[c];
  float h = 0.f;
  const int l0 = ch * TCHUNK;
  for (int t = 0; t < TCHUNK; ++t) {
    const size_t row = (size_t)b * LSEQ + l0 + t;
    h = a * h + be * xn[row * DIMC + c];
    hs[row * ACTC + c] = h;
  }
  carryB[(size_t)blockIdx.x * ACTC + c] = h;
}

__global__ __launch_bounds__(256) void scan_combine_kernel(
    const float* __restrict__ alpha, const float* __restrict__ carryB,
    float* __restrict__ carryI) {
  const int b = blockIdx.x;
  const int c = threadIdx.x;
  const float a = alpha[c];
  float aT = a;
#pragma unroll
  for (int i = 0; i < 7; ++i) aT *= aT;  // a^128
  float h = 0.f;
  for (int ch = 0; ch < NCHUNK; ++ch) {
    const size_t idx = ((size_t)b * NCHUNK + ch) * ACTC + c;
    carryI[idx] = h;                 // carry entering this chunk
    h = aT * h + carryB[idx];
  }
}

__global__ __launch_bounds__(256) void scan_fixup_kernel(
    const float* __restrict__ alpha, const float* __restrict__ carryI,
    float* __restrict__ hs) {
  const int b = blockIdx.x >> 4;
  const int ch = blockIdx.x & (NCHUNK - 1);
  const int c = threadIdx.x;
  const float a = alpha[c];
  const float cin = carryI[(size_t)blockIdx.x * ACTC + c];
  float pfac = a;
  const int l0 = ch * TCHUNK;
  for (int t = 0; t < TCHUNK; ++t) {
    const size_t row = (size_t)b * LSEQ + l0 + t;
    hs[row * ACTC + c] += pfac * cin;
    pfac *= a;
  }
}

// ------------------------------ WMMA GEMM ----------------------------------
// C[m,n] = sum_k A[m,k] * W[n,k]  (both row-major bf16), f32 accumulate.
// Block tile 128x128, K-step 32, 8 waves; wave tile 32x64 = 2x4 fragments.
// ISA 16-bit A/B striping: lane<16 -> K {0..7,16..23}; lane>=16 -> {8..15,24..31}.

__device__ inline v16bf load_frag(const __bf16* rowp, int half) {
  v8bf lo = *(const v8bf*)(rowp + half * 8);
  v8bf hi = *(const v8bf*)(rowp + 16 + half * 8);
  return __builtin_shufflevector(lo, hi, 0, 1, 2, 3, 4, 5, 6, 7, 8, 9, 10, 11,
                                 12, 13, 14, 15);
}

template <int EPI>
__global__ __launch_bounds__(256) void gemm_epi_kernel(
    const __bf16* __restrict__ A, const __bf16* __restrict__ Bw,
    const float* __restrict__ bias, const float* __restrict__ aux0,
    const float* __restrict__ aux1, const __bf16* __restrict__ auxbf,
    float* __restrict__ outf, __bf16* __restrict__ outbf, int M, int N,
    int K) {
  constexpr int BM = 128, BN = 128, BK = 32, LDSW = BK + 8;  // pad: no bank conflicts
  __shared__ __bf16 As[2][BM][LDSW];
  __shared__ __bf16 Bs[2][BN][LDSW];

  const int tid = threadIdx.x;
  const int m0 = blockIdx.y * BM;
  const int n0 = blockIdx.x * BN;
  const int lrow = tid >> 1;           // staging: row 0..127
  const int lcol = (tid & 1) * 16;     // staging: 16 bf16 halves
  const int lane = tid & 31;
  const int wid = tid >> 5;
  const int wm = (wid & 3) * 32;       // wave M offset
  const int wn = (wid >> 2) * 64;      // wave N offset
  const int r = lane & 15;
  const int half = lane >> 4;

  auto stage = [&](int kt, int buf) {
    const int k0 = kt * BK;
    const __bf16* ga = A + (size_t)(m0 + lrow) * K + (k0 + lcol);
    const __bf16* gb = Bw + (size_t)(n0 + lrow) * K + (k0 + lcol);
#if USE_ASYNC_LDS && defined(__AMDGCN__)
    // gfx1250 direct global->LDS path, tracked by ASYNCcnt: no VGPR bounce.
    __builtin_amdgcn_global_load_async_to_lds_b128(
        (AS1 v4i*)ga, (AS3 v4i*)&As[buf][lrow][lcol], 0, 0);
    __builtin_amdgcn_global_load_async_to_lds_b128(
        (AS1 v4i*)(ga + 8), (AS3 v4i*)&As[buf][lrow][lcol + 8], 0, 0);
    __builtin_amdgcn_global_load_async_to_lds_b128(
        (AS1 v4i*)gb, (AS3 v4i*)&Bs[buf][lrow][lcol], 0, 0);
    __builtin_amdgcn_global_load_async_to_lds_b128(
        (AS1 v4i*)(gb + 8), (AS3 v4i*)&Bs[buf][lrow][lcol + 8], 0, 0);
#else
    *(v8bf*)(&As[buf][lrow][lcol]) = *(const v8bf*)(ga);
    *(v8bf*)(&As[buf][lrow][lcol + 8]) = *(const v8bf*)(ga + 8);
    *(v8bf*)(&Bs[buf][lrow][lcol]) = *(const v8bf*)(gb);
    *(v8bf*)(&Bs[buf][lrow][lcol + 8]) = *(const v8bf*)(gb + 8);
#endif
  };

  auto wait_stage = [&]() {
#if USE_ASYNC_LDS && defined(__AMDGCN__)
    __builtin_amdgcn_s_wait_asynccnt(0);
#endif
  };

  v8f acc[2][4] = {};

  stage(0, 0);
  wait_stage();
  __syncthreads();
  const int nk = K / BK;
  for (int kt = 0; kt < nk; ++kt) {
    const int buf = kt & 1;
    if (kt + 1 < nk) stage(kt + 1, buf ^ 1);
    v16bf af[2], bfr[4];
#pragma unroll
    for (int i = 0; i < 2; ++i)
      af[i] = load_frag(&As[buf][wm + i * 16 + r][0], half);
#pragma unroll
    for (int j = 0; j < 4; ++j)
      bfr[j] = load_frag(&Bs[buf][wn + j * 16 + r][0], half);
#pragma unroll
    for (int i = 0; i < 2; ++i)
#pragma unroll
      for (int j = 0; j < 4; ++j)
        acc[i][j] = __builtin_amdgcn_wmma_f32_16x16x32_bf16(
            false, af[i], false, bfr[j], (short)0, acc[i][j], false, false);
    wait_stage();   // own async copies done before publishing via barrier
    __syncthreads();
  }

  // Epilogue: C element e of frag (i,j) -> row = wm+i*16+half*8+e, col = wn+j*16+r
#pragma unroll
  for (int i = 0; i < 2; ++i) {
#pragma unroll
    for (int j = 0; j < 4; ++j) {
#pragma unroll
      for (int e = 0; e < 8; ++e) {
        const int m = m0 + wm + i * 16 + half * 8 + e;
        const int n = n0 + wn + j * 16 + r;
        const size_t idx = (size_t)m * N + n;
        float v = acc[i][j][e] + bias[n];
        if constexpr (EPI == 0) {  // y = conv_gemm + bias + x_rec ; emit f32+bf16
          v += (n < ACTC) ? aux0[(size_t)m * ACTC + n]
                          : aux1[(size_t)m * DIMC + n];
          outf[idx] = v;
          outbf[idx] = (__bf16)v;
        } else if constexpr (EPI == 1) {  // sigmoid gate -> bf16
          outbf[idx] = (__bf16)(1.0f / (1.0f + __expf(-v)));
        } else if constexpr (EPI == 2) {  // y2 = y + sig*tanh(.) ; f32+bf16
          const float g = aux0[idx] + (float)auxbf[idx] * tanhf(v);
          outf[idx] = g;
          outbf[idx] = (__bf16)g;
        } else if constexpr (EPI == 3) {  // exact GELU -> bf16
          outbf[idx] = (__bf16)(0.5f * v * (1.0f + erff(v * 0.70710678118f)));
        } else {  // EPI == 4: out = up_gemm + bias + y2 + residual(x)
          outf[idx] = v + aux0[idx] + aux1[idx];
        }
      }
    }
  }
}

// ------------------------------- launcher ----------------------------------

extern "C" void kernel_launch(void* const* d_in, const int* in_sizes, int n_in,
                              void* d_out, int out_size, void* d_ws,
                              size_t ws_size, hipStream_t stream) {
  (void)in_sizes; (void)n_in; (void)out_size; (void)ws_size;
  const float* x      = (const float*)d_in[0];
  const float* norm_w = (const float*)d_in[1];
  const float* dw_w   = (const float*)d_in[2];
  const float* dw_b   = (const float*)d_in[3];
  const float* pw_w   = (const float*)d_in[4];
  const float* pw_b   = (const float*)d_in[5];
  const float* alpha  = (const float*)d_in[6];
  const float* beta   = (const float*)d_in[7];
  const float* W1_w   = (const float*)d_in[8];
  const float* W1_b   = (const float*)d_in[9];
  const float* W2_w   = (const float*)d_in[10];
  const float* W2_b   = (const float*)d_in[11];
  const float* down_w = (const float*)d_in[12];
  const float* down_b = (const float*)d_in[13];
  const float* up_w   = (const float*)d_in[14];
  const float* up_b   = (const float*)d_in[15];

  char* p = (char*)d_ws;
  auto carve = [&](size_t bytes) {
    char* q = p;
    p += (bytes + 255) & ~size_t(255);
    return q;
  };
  __bf16* pw_bf   = (__bf16*)carve((size_t)DIMC * DIMC * 2);
  __bf16* W1_bf   = (__bf16*)carve((size_t)DIMC * DIMC * 2);
  __bf16* W2_bf   = (__bf16*)carve((size_t)DIMC * DIMC * 2);
  __bf16* down_bf = (__bf16*)carve((size_t)ACTC * DIMC * 2);
  __bf16* up_bf   = (__bf16*)carve((size_t)DIMC * ACTC * 2);
  float*  xn      = (float*)carve((size_t)MROWS * DIMC * 4);
  float*  hs      = (float*)carve((size_t)MROWS * ACTC * 4);
  __bf16* xc_bf   = (__bf16*)carve((size_t)MROWS * DIMC * 2);
  float*  y_f     = (float*)carve((size_t)MROWS * DIMC * 4);
  __bf16* y_bf    = (__bf16*)carve((size_t)MROWS * DIMC * 2);
  __bf16* y2_bf   = (__bf16*)carve((size_t)MROWS * DIMC * 2);
  float*  carryB  = (float*)carve((size_t)8 * NCHUNK * ACTC * 4);
  float*  carryI  = (float*)carve((size_t)8 * NCHUNK * ACTC * 4);
  // Buffer reuse on dead ranges (stream-ordered, so safe across kernels):
  float*  y2_f  = xn;              // xn last read in GEMM<0> epilogue
  __bf16* sg_bf = xc_bf;           // xc last read as A of GEMM<0>
  __bf16* t_bf  = (__bf16*)hs;     // hs last read in GEMM<0> epilogue (8MB <= 16MB)

  // weight conversion (tiny, amortized over the 120 GFLOP of GEMM)
  cvt_bf16_kernel<<<(DIMC * DIMC + 255) / 256, 256, 0, stream>>>(pw_w, pw_bf, DIMC * DIMC);
  cvt_bf16_kernel<<<(DIMC * DIMC + 255) / 256, 256, 0, stream>>>(W1_w, W1_bf, DIMC * DIMC);
  cvt_bf16_kernel<<<(DIMC * DIMC + 255) / 256, 256, 0, stream>>>(W2_w, W2_bf, DIMC * DIMC);
  cvt_bf16_kernel<<<(ACTC * DIMC + 255) / 256, 256, 0, stream>>>(down_w, down_bf, ACTC * DIMC);
  cvt_bf16_kernel<<<(DIMC * ACTC + 255) / 256, 256, 0, stream>>>(up_w, up_bf, DIMC * ACTC);

  rmsnorm_kernel<<<MROWS, 256, 0, stream>>>(x, norm_w, xn);
  dwconv_kernel<<<MROWS, 256, 0, stream>>>(xn, dw_w, dw_b, xc_bf);

  scan_chunk_kernel<<<8 * NCHUNK, 256, 0, stream>>>(xn, alpha, beta, hs, carryB);
  scan_combine_kernel<<<8, 256, 0, stream>>>(alpha, carryB, carryI);
  scan_fixup_kernel<<<8 * NCHUNK, 256, 0, stream>>>(alpha, carryI, hs);

  dim3 blk(256);
  dim3 g1024(DIMC / 128, MROWS / 128);
  dim3 g256(ACTC / 128, MROWS / 128);
  // y = xc @ pw^T + pw_b + x_rec
  gemm_epi_kernel<0><<<g1024, blk, 0, stream>>>(xc_bf, pw_bf, pw_b, hs, xn,
                                                nullptr, y_f, y_bf, MROWS, DIMC, DIMC);
  // sgate = sigmoid(y @ W1^T + b1)
  gemm_epi_kernel<1><<<g1024, blk, 0, stream>>>(y_bf, W1_bf, W1_b, nullptr, nullptr,
                                                nullptr, nullptr, sg_bf, MROWS, DIMC, DIMC);
  // y2 = y + sgate * tanh(y @ W2^T + b2)
  gemm_epi_kernel<2><<<g1024, blk, 0, stream>>>(y_bf, W2_bf, W2_b, y_f, nullptr,
                                                sg_bf, y2_f, y2_bf, MROWS, DIMC, DIMC);
  // t = gelu(y2 @ down^T + db)
  gemm_epi_kernel<3><<<g256, blk, 0, stream>>>(y2_bf, down_bf, down_b, nullptr, nullptr,
                                               nullptr, nullptr, t_bf, MROWS, ACTC, DIMC);
  // out = y2 + t @ up^T + ub + residual(x)
  gemm_epi_kernel<4><<<g1024, blk, 0, stream>>>(t_bf, up_bf, up_b, y2_f, x, nullptr,
                                                (float*)d_out, nullptr, MROWS, DIMC, ACTC);
}
